// Model_51505247813942
// MI455X (gfx1250) — matrix operand order repaired and checked by reading, not compiled, not run on hardware
//
#include <hip/hip_runtime.h>
#include <hip/hip_bf16.h>

typedef __attribute__((ext_vector_type(16))) _Float16 v16h;
typedef __attribute__((ext_vector_type(8)))  float    v8f;

#define NNODES 100000
#define NEDGES 1600000
#define DFEAT  128
#define NTILES (NNODES / 16)   // 6250

// ---------------- workspace zeroing ----------------
__global__ void k_zero(float* __restrict__ p, long n) {
    long i = (long)blockIdx.x * blockDim.x + threadIdx.x;
    long stride = (long)gridDim.x * blockDim.x;
    for (; i < n; i += stride) p[i] = 0.0f;
}

// ---------------- in-degree counts (f32 atomics, exact below 2^24) ----------------
__global__ void k_degree(const int* __restrict__ dst_f, const int* __restrict__ dst_b,
                         float* __restrict__ degf, float* __restrict__ degb) {
    int i = blockIdx.x * blockDim.x + threadIdx.x;
    if (i < NEDGES) {
        atomicAdd(&degf[dst_f[i]], 1.0f);
    } else {
        int j = i - NEDGES;
        if (j < NEDGES) atomicAdd(&degb[dst_b[j]], 1.0f);
    }
}

// ---------------- edge scatter: one wave per edge, 4 floats per lane ----------------
__global__ void k_scatter(const float* __restrict__ x,
                          const int* __restrict__ src_f, const int* __restrict__ dst_f,
                          const int* __restrict__ src_b, const int* __restrict__ dst_b,
                          float* __restrict__ aggf, float* __restrict__ aggb) {
    long gid  = (long)blockIdx.x * blockDim.x + threadIdx.x;
    long wave = gid >> 5;
    int  lane = (int)(gid & 31);

    const int* src; const int* dst; float* agg; long e;
    if (wave < NEDGES) {
        e = wave; src = src_f; dst = dst_f; agg = aggf;
    } else {
        e = wave - NEDGES;
        if (e >= NEDGES) return;
        src = src_b; dst = dst_b; agg = aggb;
    }
    int s = src[e];
    int d = dst[e];
    const float4 v = *reinterpret_cast<const float4*>(x + (long)s * DFEAT + lane * 4);
    float* o = agg + (long)d * DFEAT + lane * 4;
    atomicAdd(o + 0, v.x);
    atomicAdd(o + 1, v.y);
    atomicAdd(o + 2, v.z);
    atomicAdd(o + 3, v.w);
}

// ---------------- weights: transpose + f32->f16, layout wt[src][n][k] ----------------
__global__ void k_wt(const float* __restrict__ relW, const float* __restrict__ loopW,
                     _Float16* __restrict__ wt) {
    int i = blockIdx.x * blockDim.x + threadIdx.x;  // over 3*128*128, i = r*16384 + n*128 + k
    if (i >= 3 * DFEAT * DFEAT) return;
    int r   = i / (DFEAT * DFEAT);
    int rem = i % (DFEAT * DFEAT);
    int n   = rem / DFEAT;   // output feature
    int k   = rem % DFEAT;   // input feature
    float w = (r < 2) ? relW[(long)r * DFEAT * DFEAT + (long)k * DFEAT + n]
                      : loopW[(long)k * DFEAT + n];
    wt[i] = (_Float16)w;
}

// ---------------- fused normalize + 3-source GEMM + bias + ReLU ----------------
// One wave -> 16 rows x 128 cols of output. K = 3 sources x 128.
// B (weights) staged per-source in LDS (32 KB), shared by the block's 8 waves.
__global__ __launch_bounds__(256) void k_gemm(
    const float* __restrict__ aggf, const float* __restrict__ aggb,
    const float* __restrict__ x,
    const float* __restrict__ degf, const float* __restrict__ degb,
    const _Float16* __restrict__ wt, const float* __restrict__ bias,
    float* __restrict__ out) {

    __shared__ __align__(32) _Float16 lwt[DFEAT * DFEAT];   // 32 KB: weights of current source

    const int wave = threadIdx.x >> 5;
    const int lane = threadIdx.x & 31;
    int tile = blockIdx.x * 8 + wave;
    if (tile >= NTILES) tile = NTILES - 1;     // clamp: surplus waves redo last tile
                                               // (identical values; keeps barriers/EXEC uniform)

    const int m0      = tile * 16;
    const int mA      = m0 + (lane & 15);      // the A-matrix row this lane serves
    const int halfsel = lane >> 4;             // 0: lanes 0-15, 1: lanes 16-31
    const int kbaseA  = halfsel * 16;          // A k-offset within each 32-wide k-step
    const int kbaseB  = halfsel * 16;          // B k-offset within each 32-wide k-step

    v8f acc[8];
    #pragma unroll
    for (int j = 0; j < 8; ++j) acc[j] = (v8f){};

    const float* srcs[3] = { aggf, aggb, x };
    float scales[3];
    scales[0] = __builtin_amdgcn_rcpf(fmaxf(degf[mA], 1.0f));  // right-norm, deg clamp(min=1)
    scales[1] = __builtin_amdgcn_rcpf(fmaxf(degb[mA], 1.0f));
    scales[2] = 1.0f;

    #pragma unroll
    for (int s = 0; s < 3; ++s) {
        // cooperative LDS stage of this source's 128x128 f16 weights (2048 uint4)
        {
            const uint4* g = reinterpret_cast<const uint4*>(wt + (long)s * DFEAT * DFEAT);
            uint4* l = reinterpret_cast<uint4*>(lwt);
            #pragma unroll
            for (int t = 0; t < 8; ++t)
                l[threadIdx.x + 256 * t] = g[threadIdx.x + 256 * t];
        }
        __syncthreads();

        const float* rowp = srcs[s] + (long)mA * DFEAT;
        const float  sc   = scales[s];
        #pragma unroll
        for (int ks = 0; ks < 4; ++ks) {
            const int kk = ks * 32;
            // A lane layout (16-bit A 16x32): elems 0..15 -> K=kbase+0..15 (contiguous, matches B)
            const float4 q0 = *reinterpret_cast<const float4*>(rowp + kk + kbaseA);
            const float4 q1 = *reinterpret_cast<const float4*>(rowp + kk + kbaseA + 4);
            const float4 q2 = *reinterpret_cast<const float4*>(rowp + kk + kbaseA + 8);
            const float4 q3 = *reinterpret_cast<const float4*>(rowp + kk + kbaseA + 12);
            v16h a;
            a[0]  = (_Float16)(q0.x * sc); a[1]  = (_Float16)(q0.y * sc);
            a[2]  = (_Float16)(q0.z * sc); a[3]  = (_Float16)(q0.w * sc);
            a[4]  = (_Float16)(q1.x * sc); a[5]  = (_Float16)(q1.y * sc);
            a[6]  = (_Float16)(q1.z * sc); a[7]  = (_Float16)(q1.w * sc);
            a[8]  = (_Float16)(q2.x * sc); a[9]  = (_Float16)(q2.y * sc);
            a[10] = (_Float16)(q2.z * sc); a[11] = (_Float16)(q2.w * sc);
            a[12] = (_Float16)(q3.x * sc); a[13] = (_Float16)(q3.y * sc);
            a[14] = (_Float16)(q3.z * sc); a[15] = (_Float16)(q3.w * sc);

            #pragma unroll
            for (int j = 0; j < 8; ++j) {
                // B lane layout (16-bit B 32x16): lane = column n, 16 contiguous K per half-wave
                const int n = j * 16 + (lane & 15);
                const v16h b = *reinterpret_cast<const v16h*>(lwt + n * DFEAT + kk + kbaseB);
                acc[j] = __builtin_amdgcn_wmma_f32_16x16x32_f16(
                    /*neg_a=*/false, a, /*neg_b=*/false, b,
                    /*c_mod=*/(short)0, acc[j], /*reuse_a=*/false, /*reuse_b=*/false);
            }
        }
        __syncthreads();   // protect lwt before next source overwrites it
    }

    // Epilogue: C/D layout -> VGPR r holds row m0 + r + 8*halfsel, col = 16j + (lane&15)
    const int mrow = m0 + 8 * halfsel;
    #pragma unroll
    for (int j = 0; j < 8; ++j) {
        const int n = j * 16 + (lane & 15);
        const float bv = bias[n];
        #pragma unroll
        for (int r = 0; r < 8; ++r) {
            float v = acc[j][r] + bv;
            v = fmaxf(v, 0.0f);
            out[(long)(mrow + r) * DFEAT + n] = v;
        }
    }
}

extern "C" void kernel_launch(void* const* d_in, const int* in_sizes, int n_in,
                              void* d_out, int out_size, void* d_ws, size_t ws_size,
                              hipStream_t stream) {
    (void)in_sizes; (void)n_in; (void)out_size; (void)ws_size;

    const float* x     = (const float*)d_in[0];
    const int*   src_f = (const int*)d_in[1];
    const int*   dst_f = (const int*)d_in[2];
    const int*   src_b = (const int*)d_in[3];
    const int*   dst_b = (const int*)d_in[4];
    const float* relW  = (const float*)d_in[5];
    const float* loopW = (const float*)d_in[6];
    const float* bias  = (const float*)d_in[7];
    float* out = (float*)d_out;

    // ws layout (bytes): aggf | aggb | degf | degb | wt16   (~103.3 MB total)
    char* ws = (char*)d_ws;
    const size_t AGG_BYTES = (size_t)NNODES * DFEAT * sizeof(float);  // 51.2 MB
    const size_t DEG_BYTES = (size_t)NNODES * sizeof(float);          // 0.4 MB
    float*    aggf = (float*)(ws);
    float*    aggb = (float*)(ws + AGG_BYTES);
    float*    degf = (float*)(ws + 2 * AGG_BYTES);
    float*    degb = (float*)(ws + 2 * AGG_BYTES + DEG_BYTES);
    _Float16* wt   = (_Float16*)(ws + 2 * AGG_BYTES + 2 * DEG_BYTES);

    // 1) zero agg + deg (contiguous region)
    const long zeroN = (long)2 * NNODES * DFEAT + (long)2 * NNODES;
    k_zero<<<2048, 256, 0, stream>>>(aggf, zeroN);

    // 2) degrees for both relations
    const int degThreads = 2 * NEDGES;
    k_degree<<<(degThreads + 255) / 256, 256, 0, stream>>>(dst_f, dst_b, degf, degb);

    // 3) edge scatter: one wave per edge, both relations
    const long scatThreads = (long)2 * NEDGES * 32;
    k_scatter<<<(unsigned)((scatThreads + 255) / 256), 256, 0, stream>>>(
        x, src_f, dst_f, src_b, dst_b, aggf, aggb);

    // 4) weights -> f16, transposed to [src][n][k]
    k_wt<<<(3 * DFEAT * DFEAT + 255) / 256, 256, 0, stream>>>(relW, loopW, wt);

    // 5) fused normalize + GEMM + bias + ReLU (6250 row-tiles, 8 waves/block)
    k_gemm<<<(NTILES + 7) / 8, 256, 0, stream>>>(aggf, aggb, x, degf, degb, wt, bias, out);
}
